// LSTM_76828374990974
// MI455X (gfx1250) — compile-verified, matrix operand order checked by reading, not run on hardware
//
#include <hip/hip_runtime.h>
#include <hip/hip_bf16.h>

// ---------------------------------------------------------------------------
// 2-layer LSTM over [B=64, T=48, N=1024, D=64], H=64.
// Rows (B*N = 65536) independent; T sequential. Compute-bound (~412 GFLOP vs
// ~850 MB traffic, AI ~485 flop/B) -> bf16 WMMA with f32 accumulate.
// Weights transposed+bf16 in LDS; h recirculates through per-wave LDS tiles
// (does the C-layout -> A-layout transpose); c-state/biases in VGPRs.
// Activations use v_exp_f32 + v_rcp_f32 (no IEEE divide sequences).
// ---------------------------------------------------------------------------

#define B_ 64
#define T_ 48
#define N_ 1024
#define D_ 64
#define H_ 64

typedef __attribute__((ext_vector_type(16))) __bf16 bf16x16;
typedef __attribute__((ext_vector_type(8)))  __bf16 bf16x8;
typedef __attribute__((ext_vector_type(8)))  float  f32x8;
typedef __attribute__((ext_vector_type(4)))  float  f32x4;

__device__ __forceinline__ float fast_rcp(float x) {
  return __builtin_amdgcn_rcpf(x);          // v_rcp_f32, 1 op
}
__device__ __forceinline__ float fast_sig(float x) {
  return fast_rcp(1.0f + __expf(-x));       // v_exp + v_rcp
}
__device__ __forceinline__ float fast_tanh(float x) {
  return 1.0f - 2.0f * fast_rcp(__expf(2.0f * x) + 1.0f);
}

// Load a 16x32 bf16 WMMA A/B fragment from a row-major bf16 matrix in LDS.
// ISA layout (16-bit A 16x32): lanes 0-15 hold K=0..7 (v0-3) and K=16..23
// (v4-7) of row M=lane; lanes 16-31 hold K=8..15 and K=24..31.
__device__ __forceinline__ bf16x16 load_frag_lds(const __bf16* base, int stride,
                                                 int lane) {
  int r  = lane & 15;
  int h8 = (lane >> 4) << 3;  // 0 or 8
  const __bf16* p = base + r * stride + h8;
  bf16x8 lo = *(const bf16x8*)(p);        // K = h8    .. h8+7
  bf16x8 hi = *(const bf16x8*)(p + 16);   // K = 16+h8 .. 16+h8+7
  bf16x16 out;
#pragma unroll
  for (int i = 0; i < 8; ++i) { out[i] = lo[i]; out[i + 8] = hi[i]; }
  return out;
}

// Same fragment, sourced from global f32 (converted to bf16 in-register).
__device__ __forceinline__ bf16x16 load_frag_global_f32(const float* base,
                                                        int stride, int lane) {
  int r  = lane & 15;
  int h8 = (lane >> 4) << 3;
  const float* p = base + (size_t)r * stride + h8;
  f32x4 a0 = *(const f32x4*)(p);
  f32x4 a1 = *(const f32x4*)(p + 4);
  f32x4 b0 = *(const f32x4*)(p + 16);
  f32x4 b1 = *(const f32x4*)(p + 20);
  bf16x16 out;
#pragma unroll
  for (int i = 0; i < 4; ++i) {
    out[i]      = (__bf16)a0[i];
    out[i + 4]  = (__bf16)a1[i];
    out[i + 8]  = (__bf16)b0[i];
    out[i + 12] = (__bf16)b1[i];
  }
  return out;
}

// One LSTM cell update for a wave's 16 rows.
__device__ __forceinline__ void lstm_layer(const __bf16* Wt, const bf16x16 A[4],
                                           const float bias[16], float c[32],
                                           __bf16* hbuf, float* outp,
                                           int lane) {
  int colL = lane & 15;
  int hi8  = (lane >> 4) << 3;
#pragma unroll
  for (int cc = 0; cc < 4; ++cc) {   // 16-column chunk of H
    f32x8 acc[4];                    // i, f, g, o fragments (aligned layouts)
#pragma unroll
    for (int g = 0; g < 4; ++g) {
      float bv = bias[g * 4 + cc];
#pragma unroll
      for (int e = 0; e < 8; ++e) acc[g][e] = bv;
    }
#pragma unroll
    for (int k = 0; k < 4; ++k) {    // K = 128 in steps of 32
      // batch the 4 gate B-fragment loads so they overlap prior WMMAs
      bf16x16 bw[4];
#pragma unroll
      for (int g = 0; g < 4; ++g)
        bw[g] = load_frag_lds(Wt + (g * 4 + cc) * 16 * 128 + k * 32, 128, lane);
#pragma unroll
      for (int g = 0; g < 4; ++g)
        acc[g] = __builtin_amdgcn_wmma_f32_16x16x32_bf16(
            false, A[k], false, bw[g], (short)0, acc[g], false, false);
    }
    // element-wise cell update; C-fragment element e maps to (row e+hi8, colL)
#pragma unroll
    for (int e = 0; e < 8; ++e) {
      float iv = fast_sig(acc[0][e]);
      float fv = fast_sig(acc[1][e]);
      float gv = fast_tanh(acc[2][e]);
      float ov = fast_sig(acc[3][e]);
      float cy = fv * c[cc * 8 + e] + iv * gv;
      float hy = ov * fast_tanh(cy);
      c[cc * 8 + e] = cy;
      int row = e + hi8;
      hbuf[row * H_ + cc * 16 + colL] = (__bf16)hy;
      if (outp) outp[row * H_ + cc * 16 + colL] = hy;
    }
  }
}

__global__ __launch_bounds__(256, 1)
void lstm2_wmma_kernel(const float* __restrict__ x,
                       const float* __restrict__ W0,
                       const float* __restrict__ b0,
                       const float* __restrict__ W1,
                       const float* __restrict__ b1,
                       float* __restrict__ out) {
  extern __shared__ __align__(16) char smem[];
  __bf16* Wt0 = (__bf16*)smem;             // [256][128] bf16 = 64 KB
  __bf16* Wt1 = Wt0 + 256 * 128;           // 64 KB
  __bf16* hb  = Wt1 + 256 * 128;           // 8 waves x 2 layers x 16x64 = 32 KB

  const int tid  = threadIdx.x;
  const int lane = tid & 31;
  const int wave = tid >> 5;

  // Stage transposed bf16 weights: Wt[n][k] = W[k][n] (one-time, L2-resident).
  for (int i = tid; i < 256 * 128; i += 256) {
    int n = i >> 7, k = i & 127;
    Wt0[i] = (__bf16)W0[k * 256 + n];
    Wt1[i] = (__bf16)W1[k * 256 + n];
  }

  __bf16* hbuf0 = hb + wave * (2 * 16 * H_);
  __bf16* hbuf1 = hbuf0 + 16 * H_;
  for (int i = lane; i < 2 * 16 * H_; i += 32) hbuf0[i] = (__bf16)0.0f;

  // Per-lane gate biases (column-dependent only).
  float bias0[16], bias1[16];
  {
    int colL = lane & 15;
#pragma unroll
    for (int g = 0; g < 4; ++g)
#pragma unroll
      for (int cc = 0; cc < 4; ++cc) {
        bias0[g * 4 + cc] = b0[g * 64 + cc * 16 + colL];
        bias1[g * 4 + cc] = b1[g * 64 + cc * 16 + colL];
      }
  }

  float c0[32], c1[32];
#pragma unroll
  for (int i = 0; i < 32; ++i) { c0[i] = 0.0f; c1[i] = 0.0f; }

  __syncthreads();

  // Wave's 16 rows are 16 consecutive n within one b (128 | N).
  const int rowBase = blockIdx.x * 128 + wave * 16;
  const int bIdx = rowBase >> 10;          // / N_
  const int n0   = rowBase & (N_ - 1);
  const float* xb  = x + (((size_t)bIdx * T_) * N_ + n0) * D_;
  float* outw      = out + (size_t)rowBase * H_;

#pragma unroll 1   // body is internally unrolled; keep the serial loop tight
  for (int t = 0; t < T_; ++t) {
    const float* xt = xb + (size_t)t * N_ * D_;

    // ---- layer 0: A = [x_t | h0_prev] ----
    // (same-wave LDS in-order execution covers the WAR on hbuf0 here)
    bf16x16 A0[4];
    A0[0] = load_frag_global_f32(xt,      D_, lane);  // K  0..31
    A0[1] = load_frag_global_f32(xt + 32, D_, lane);  // K 32..63
    A0[2] = load_frag_lds(hbuf0,      H_, lane);      // K 64..95  (h0)
    A0[3] = load_frag_lds(hbuf0 + 32, H_, lane);      // K 96..127 (h0)
    lstm_layer(Wt0, A0, bias0, c0, hbuf0, nullptr, lane);
    __syncthreads();  // cross-lane RAW: h0_new -> layer-1 A fragments

    // ---- layer 1: A = [h0_new | h1_prev] ----
    bf16x16 A1[4];
    A1[0] = load_frag_lds(hbuf0,      H_, lane);
    A1[1] = load_frag_lds(hbuf0 + 32, H_, lane);
    A1[2] = load_frag_lds(hbuf1,      H_, lane);
    A1[3] = load_frag_lds(hbuf1 + 32, H_, lane);
    lstm_layer(Wt1, A1, bias1, c1, hbuf1,
               (t == T_ - 1) ? outw : nullptr, lane);
    __syncthreads();  // cross-lane RAW: h1_new / h0_new -> next timestep
  }
}

extern "C" void kernel_launch(void* const* d_in, const int* in_sizes, int n_in,
                              void* d_out, int out_size, void* d_ws,
                              size_t ws_size, hipStream_t stream) {
  (void)in_sizes; (void)n_in; (void)d_ws; (void)ws_size; (void)out_size;
  const float* x  = (const float*)d_in[0];
  const float* W0 = (const float*)d_in[1];
  const float* b0 = (const float*)d_in[2];
  const float* W1 = (const float*)d_in[3];
  const float* b1 = (const float*)d_in[4];
  float* out = (float*)d_out;

  const size_t smem = (size_t)(2 * 256 * 128 + 8 * 2 * 16 * H_) * sizeof(__bf16);
  static bool attr_set = false;
  if (!attr_set) {
    hipFuncSetAttribute((const void*)lstm2_wmma_kernel,
                        hipFuncAttributeMaxDynamicSharedMemorySize, (int)smem);
    attr_set = true;
  }
  const int rows = B_ * N_;               // 65536
  dim3 grid(rows / 128), block(256);      // 8 waves/block, 16 rows/wave
  lstm2_wmma_kernel<<<grid, block, smem, stream>>>(x, W0, b0, W1, b1, out);
}
